// SALAD_88905823027240
// MI455X (gfx1250) — matrix operand order repaired and checked by reading, not compiled, not run on hardware
//
#include <hip/hip_runtime.h>

#define DEV __device__ __forceinline__

typedef __attribute__((ext_vector_type(16))) __bf16 v16bf;
typedef __attribute__((ext_vector_type(8)))  float  v8f;

union FragCast { uint4 u[2]; v16bf v; };
union H8Cast   { unsigned short h[8]; uint4 u; };

// ---- constants for this problem ----
#define BB   32
#define CC   1536
#define NN   1024      // H*W
#define M1   1024      // fused c1+s1 out channels
#define DD   128       // cluster feature dim
#define KK   64        // clusters

DEV unsigned short f2bf(float f) {
    unsigned u = __float_as_uint(f);
    u += 0x7FFFu + ((u >> 16) & 1u);   // round-to-nearest-even
    return (unsigned short)(u >> 16);
}

// A fragment: 16x32 bf16 tile, rows row0..row0+15, k0..k0+31, A row-major [M][lda]
// ISA layout: lanes 0-15 -> M=lane, K in {h*8+0..7, 16+h*8+0..7} with h = lane>>4
DEV v16bf load_a_frag(const unsigned short* A, int lda, int row0, int k0, int lane) {
    const int r = lane & 15, h = lane >> 4;
    const unsigned short* p = A + (size_t)(row0 + r) * lda + k0 + h * 8;
    FragCast t;
    t.u[0] = *reinterpret_cast<const uint4*>(p);       // K = h*8 + 0..7
    t.u[1] = *reinterpret_cast<const uint4*>(p + 16);  // K = 16 + h*8 + 0..7
    return t.v;
}

// B fragment: 32x16 bf16 tile (KxN'), from B^T storage [col][ldb] (K contiguous per column)
// ISA layout: lanes 0-15 -> N=lane, K in {h*8+0..7, 16+h*8+0..7} with h = lane>>4
DEV v16bf load_b_frag(const unsigned short* Bt, int ldb, int col0, int k0, int lane) {
    const int c = lane & 15, h = lane >> 4;
    const unsigned short* p = Bt + (size_t)(col0 + c) * ldb + k0 + h * 8;
    FragCast t;
    t.u[0] = *reinterpret_cast<const uint4*>(p);       // K = h*8 + 0..7
    t.u[1] = *reinterpret_cast<const uint4*>(p + 16);  // K = 16 + h*8 + 0..7
    return t.v;
}

DEV v8f wmma_bf16(v16bf a, v16bf b, v8f c) {
    return __builtin_amdgcn_wmma_f32_16x16x32_bf16(false, a, false, b, (short)0, c, false, false);
}

// ---------------- x transpose + fp32->bf16 : f_map[b][c][n] -> xT[b][n][c] ----------------
__global__ __launch_bounds__(256) void convert_x_kernel(const float* __restrict__ fmap,
                                                        unsigned short* __restrict__ xT) {
    __shared__ float tile[32][33];
    const int b = blockIdx.z;
    const int c0 = blockIdx.x * 32, n0 = blockIdx.y * 32;
    const int tx = threadIdx.x & 31, ty = threadIdx.x >> 5; // ty: 0..7
    const float* src = fmap + ((size_t)b * CC + c0) * NN + n0;
#pragma unroll
    for (int i = 0; i < 4; i++)
        tile[ty + 8 * i][tx] = src[(size_t)(ty + 8 * i) * NN + tx];
    __syncthreads();
    unsigned short* dst = xT + ((size_t)b * NN + n0) * CC + c0;
#pragma unroll
    for (int i = 0; i < 4; i++)
        dst[(size_t)(ty + 8 * i) * CC + tx] = f2bf(tile[tx][ty + 8 * i]);
}

// ---------------- weight prep ----------------
__global__ void pack_w1_kernel(const float* __restrict__ wc1, const float* __restrict__ ws1,
                               unsigned short* __restrict__ w1) {
    int i = blockIdx.x * 256 + threadIdx.x;          // i < 1024*1536, rows 0..511 = c1
    if (i < M1 * CC)
        w1[i] = f2bf(i < 512 * CC ? wc1[i] : ws1[i - 512 * CC]);
}
__global__ void pack_b1_kernel(const float* __restrict__ bc1, const float* __restrict__ bs1,
                               float* __restrict__ b1) {
    int i = blockIdx.x * 256 + threadIdx.x;
    if (i < M1) b1[i] = i < 512 ? bc1[i] : bs1[i - 512];
}
__global__ void f32_to_bf16_kernel(const float* __restrict__ s, unsigned short* __restrict__ d, int n) {
    int i = blockIdx.x * 256 + threadIdx.x;
    if (i < n) d[i] = f2bf(s[i]);
}

// ---------------- generic WMMA GEMM ----------------
// D[m][n'] = A[m][:] . B^T[n'][:]  (K inner). Block tile 64(M) x 256(N'),
// wave tile 32(M) x 64(N') = 2x4 WMMA tiles, software-pipelined (double-buffered frags).
// Outputs (any may be null):
//   outTf : f32, transposed [n'][ldmf]   (n'-major)
//   outTb : bf16, transposed [n'][ldmb]
//   outMN : bf16, natural   [m][ldn]
__global__ __launch_bounds__(256) void wmma_gemm_kernel(
    const unsigned short* __restrict__ A, unsigned long long aStride, int lda,
    const float* __restrict__ bias,
    const unsigned short* __restrict__ B, unsigned long long bStride, int ldb, int kOff,
    int K, int relu, int Nlimit,
    float* __restrict__ outTf, unsigned long long ofStride, int ldmf,
    unsigned short* __restrict__ outTb, unsigned long long obStride, int ldmb,
    unsigned short* __restrict__ outMN, unsigned long long omnStride, int ldn) {
    const int lane = threadIdx.x & 31;
    const int w    = threadIdx.x >> 5;
    const int m0   = blockIdx.x * 64 + (w & 1) * 32;
    const int n0   = blockIdx.y * 256 + (w >> 1) * 64;
    if (n0 >= Nlimit) return;          // wave-uniform: whole waves retire, EXEC stays all-1s
    const int z = blockIdx.z;
    const unsigned short* Ab = A + (size_t)z * aStride;
    const unsigned short* Bb = B + (size_t)z * bStride + kOff;

    v8f acc[2][4];
#pragma unroll
    for (int i = 0; i < 2; i++)
#pragma unroll
        for (int j = 0; j < 4; j++)
#pragma unroll
            for (int r = 0; r < 8; r++) acc[i][j][r] = 0.0f;

    // prologue: K-tile 0 in flight
    v16bf a0 = load_a_frag(Ab, lda, m0,      0, lane);
    v16bf a1 = load_a_frag(Ab, lda, m0 + 16, 0, lane);
    v16bf b0 = load_b_frag(Bb, ldb, n0 +  0, 0, lane);
    v16bf b1 = load_b_frag(Bb, ldb, n0 + 16, 0, lane);
    v16bf b2 = load_b_frag(Bb, ldb, n0 + 32, 0, lane);
    v16bf b3 = load_b_frag(Bb, ldb, n0 + 48, 0, lane);

    for (int k0 = 32; k0 < K; k0 += 32) {
        // speculative prefetch one tile ahead (global_prefetch_b8); OOB prefetch is harmless
        __builtin_prefetch(Ab + (size_t)(m0 + (lane & 15)) * lda + k0 + 32, 0, 3);
        __builtin_prefetch(Bb + (size_t)(n0 + lane) * ldb + k0 + 32, 0, 3);
        // issue next K-tile loads into fresh registers ...
        v16bf na0 = load_a_frag(Ab, lda, m0,      k0, lane);
        v16bf na1 = load_a_frag(Ab, lda, m0 + 16, k0, lane);
        v16bf nb0 = load_b_frag(Bb, ldb, n0 +  0, k0, lane);
        v16bf nb1 = load_b_frag(Bb, ldb, n0 + 16, k0, lane);
        v16bf nb2 = load_b_frag(Bb, ldb, n0 + 32, k0, lane);
        v16bf nb3 = load_b_frag(Bb, ldb, n0 + 48, k0, lane);
        // ... while the WMMAs consume the already-resident tile
        acc[0][0] = wmma_bf16(a0, b0, acc[0][0]);
        acc[1][0] = wmma_bf16(a1, b0, acc[1][0]);
        acc[0][1] = wmma_bf16(a0, b1, acc[0][1]);
        acc[1][1] = wmma_bf16(a1, b1, acc[1][1]);
        acc[0][2] = wmma_bf16(a0, b2, acc[0][2]);
        acc[1][2] = wmma_bf16(a1, b2, acc[1][2]);
        acc[0][3] = wmma_bf16(a0, b3, acc[0][3]);
        acc[1][3] = wmma_bf16(a1, b3, acc[1][3]);
        a0 = na0; a1 = na1; b0 = nb0; b1 = nb1; b2 = nb2; b3 = nb3;
    }
    // epilogue K-tile
    acc[0][0] = wmma_bf16(a0, b0, acc[0][0]);
    acc[1][0] = wmma_bf16(a1, b0, acc[1][0]);
    acc[0][1] = wmma_bf16(a0, b1, acc[0][1]);
    acc[1][1] = wmma_bf16(a1, b1, acc[1][1]);
    acc[0][2] = wmma_bf16(a0, b2, acc[0][2]);
    acc[1][2] = wmma_bf16(a1, b2, acc[1][2]);
    acc[0][3] = wmma_bf16(a0, b3, acc[0][3]);
    acc[1][3] = wmma_bf16(a1, b3, acc[1][3]);

    // bias hoisted per m-subtile (independent of n')
    float bv[2][8];
#pragma unroll
    for (int mi = 0; mi < 2; mi++)
#pragma unroll
        for (int r = 0; r < 8; r++)
            bv[mi][r] = bias ? bias[m0 + mi * 16 + (lane >> 4) * 8 + r] : 0.0f;

#pragma unroll
    for (int mi = 0; mi < 2; mi++)
#pragma unroll
        for (int ni = 0; ni < 4; ni++) {
            const int col  = n0 + ni * 16 + (lane & 15);
            const int mrow = m0 + mi * 16 + (lane >> 4) * 8;  // 8 contiguous rows per lane
            float vals[8];
#pragma unroll
            for (int r = 0; r < 8; r++) {
                float v = acc[mi][ni][r] + bv[mi][r];
                if (relu) v = fmaxf(v, 0.0f);
                vals[r] = v;
            }
            if (outTf) {
                float* p = outTf + (size_t)z * ofStride + (size_t)col * ldmf + mrow;
                *reinterpret_cast<float4*>(p)     = make_float4(vals[0], vals[1], vals[2], vals[3]);
                *reinterpret_cast<float4*>(p + 4) = make_float4(vals[4], vals[5], vals[6], vals[7]);
            }
            if (outTb) {
                unsigned short* p = outTb + (size_t)z * obStride + (size_t)col * ldmb + mrow;
                H8Cast hc;
#pragma unroll
                for (int r = 0; r < 8; r++) hc.h[r] = f2bf(vals[r]);
                *reinterpret_cast<uint4*>(p) = hc.u;
            }
            if (outMN) {
                unsigned short* p = outMN + (size_t)z * omnStride + (size_t)mrow * ldn + col;
#pragma unroll
                for (int r = 0; r < 8; r++) p[(size_t)r * ldn] = f2bf(vals[r]);
            }
        }
}

// ---------------- per-pixel L2 normalize of f: f_nd f32 [b][n][128] -> fnorm bf16 [b][n][128] ---
__global__ __launch_bounds__(256) void normf_kernel(const float* __restrict__ f_nd,
                                                    unsigned short* __restrict__ fnorm) {
    int idx = blockIdx.x * 256 + threadIdx.x;   // b*1024 + n
    const float* p = f_nd + (size_t)idx * DD;
    float s = 0.f;
#pragma unroll 8
    for (int i = 0; i < DD; i++) { float v = p[i]; s += v * v; }
    float inv = 1.0f / fmaxf(sqrtf(s), 1e-12f);
    unsigned short* q = fnorm + (size_t)idx * DD;
#pragma unroll 8
    for (int i = 0; i < DD; i++) q[i] = f2bf(p[i] * inv);
}

// ---------------- fused sim + sigmoid + row-sum: w[b][n] = sum_m sigmoid(a*sim+b) ----------------
// one wave per 16-row strip of sim; A frags loaded once, loop over all m columns.
__global__ __launch_bounds__(256) void sim_kernel(const unsigned short* __restrict__ fnorm,
                                                  float* __restrict__ wout,
                                                  const float* __restrict__ a_par,
                                                  const float* __restrict__ b_par) {
    const int lane = threadIdx.x & 31, w = threadIdx.x >> 5;
    const int gid  = blockIdx.x * 8 + w;
    const int b    = gid >> 6;
    const int n0   = (gid & 63) * 16;
    const unsigned short* F = fnorm + (size_t)b * NN * DD;
    const float ap = a_par[0], bp = b_par[0];

    v16bf A0 = load_a_frag(F, DD, n0, 0, lane);
    v16bf A1 = load_a_frag(F, DD, n0, 32, lane);
    v16bf A2 = load_a_frag(F, DD, n0, 64, lane);
    v16bf A3 = load_a_frag(F, DD, n0, 96, lane);

    v8f accsum;
#pragma unroll
    for (int r = 0; r < 8; r++) accsum[r] = 0.0f;

    for (int m0 = 0; m0 < NN; m0 += 16) {
        v16bf B0 = load_b_frag(F, DD, m0, 0, lane);
        v16bf B1 = load_b_frag(F, DD, m0, 32, lane);
        v16bf B2 = load_b_frag(F, DD, m0, 64, lane);
        v16bf B3 = load_b_frag(F, DD, m0, 96, lane);
        v8f c;
#pragma unroll
        for (int r = 0; r < 8; r++) c[r] = 0.0f;
        c = wmma_bf16(A0, B0, c);
        c = wmma_bf16(A1, B1, c);
        c = wmma_bf16(A2, B2, c);
        c = wmma_bf16(A3, B3, c);
#pragma unroll
        for (int r = 0; r < 8; r++)
            accsum[r] += 1.0f / (1.0f + expf(-(ap * c[r] + bp)));
    }
    // sum over the 16 columns held by each half-wave (masks < 16 stay in-half, wave32)
#pragma unroll
    for (int off = 1; off < 16; off <<= 1)
#pragma unroll
        for (int r = 0; r < 8; r++) accsum[r] += __shfl_xor(accsum[r], off, 32);
    if ((lane & 15) == 0) {
        float* p = wout + (size_t)b * NN + n0 + (lane >> 4) * 8;
#pragma unroll
        for (int r = 0; r < 8; r++) p[r] = accsum[r];
    }
}

// ---------------- softmax over K + divide by w^p : pmat bf16 [b][k][n] ----------------
__global__ __launch_bounds__(256) void pmat_kernel(const float* __restrict__ scores,  // [b][n][64]
                                                   const float* __restrict__ wbuf,    // [b][n]
                                                   const float* __restrict__ p_par,
                                                   unsigned short* __restrict__ pmat) {
    int idx = blockIdx.x * 256 + threadIdx.x;   // b*1024+n
    int b = idx >> 10, n = idx & (NN - 1);
    const float* s = scores + (size_t)idx * KK;
    float mx = s[0];
    for (int k = 1; k < KK; k++) mx = fmaxf(mx, s[k]);
    float sum = 0.f;
    for (int k = 0; k < KK; k++) sum += expf(s[k] - mx);
    float denom = powf(wbuf[idx], p_par[0]) + 1e-6f;
    float scale = 1.0f / (sum * denom);
    for (int k = 0; k < KK; k++)
        pmat[((size_t)b * KK + k) * NN + n] = f2bf(expf(s[k] - mx) * scale);
}

// ---------------- token MLP (tiny) ----------------
__global__ __launch_bounds__(256) void tok1_kernel(const float* __restrict__ token,
                                                   const float* __restrict__ W, const float* __restrict__ bias,
                                                   float* __restrict__ h1) {
    int b = blockIdx.x, o = blockIdx.y * 256 + threadIdx.x;
    const float* t = token + (size_t)b * CC;
    const float* w = W + (size_t)o * CC;
    float s = bias[o];
    for (int i = 0; i < CC; i++) s += t[i] * w[i];
    h1[(size_t)b * 512 + o] = fmaxf(s, 0.0f);
}
__global__ __launch_bounds__(256) void tok2_kernel(const float* __restrict__ h1,
                                                   const float* __restrict__ W, const float* __restrict__ bias,
                                                   float* __restrict__ t) {
    int b = blockIdx.x, o = threadIdx.x;
    const float* h = h1 + (size_t)b * 512;
    const float* w = W + (size_t)o * 512;
    float s = bias[o];
    for (int i = 0; i < 512; i++) s += h[i] * w[i];
    t[(size_t)b * 256 + o] = s;
}

// ---------------- final: intra-norm v, norm t, concat, global L2 norm ----------------
__global__ __launch_bounds__(256) void final_kernel(const float* __restrict__ v,   // [b][128][64]
                                                    const float* __restrict__ t,   // [b][256]
                                                    float* __restrict__ out) {     // [b][8448]
    __shared__ float y[256 + DD * KK];
    __shared__ float red[256];
    __shared__ float invk[KK];
    const int b = blockIdx.x, tid = threadIdx.x;
    const float* vb = v + (size_t)b * DD * KK;
    for (int i = tid; i < DD * KK; i += 256) y[256 + i] = vb[i];
    __syncthreads();
    if (tid < KK) {
        float s = 0.f;
        for (int d = 0; d < DD; d++) { float x = y[256 + d * KK + tid]; s += x * x; }
        invk[tid] = 1.0f / fmaxf(sqrtf(s), 1e-12f);
    }
    float tv = t[(size_t)b * 256 + tid];
    red[tid] = tv * tv;
    __syncthreads();
    for (int s = 128; s > 0; s >>= 1) { if (tid < s) red[tid] += red[tid + s]; __syncthreads(); }
    float invt = 1.0f / fmaxf(sqrtf(red[0]), 1e-12f);
    __syncthreads();
    y[tid] = tv * invt;
    for (int i = tid; i < DD * KK; i += 256) y[256 + i] *= invk[i & (KK - 1)];
    __syncthreads();
    float s = 0.f;
    for (int i = tid; i < 256 + DD * KK; i += 256) { float x = y[i]; s += x * x; }
    red[tid] = s;
    __syncthreads();
    for (int st = 128; st > 0; st >>= 1) { if (tid < st) red[tid] += red[tid + st]; __syncthreads(); }
    float invg = 1.0f / fmaxf(sqrtf(red[0]), 1e-12f);
    __syncthreads();
    float* o = out + (size_t)b * (256 + DD * KK);
    for (int i = tid; i < 256 + DD * KK; i += 256) o[i] = y[i] * invg;
}

extern "C" void kernel_launch(void* const* d_in, const int* in_sizes, int n_in,
                              void* d_out, int out_size, void* d_ws, size_t ws_size,
                              hipStream_t stream) {
    (void)in_sizes; (void)n_in; (void)out_size; (void)ws_size;
    const float* f_map = (const float*)d_in[0];
    const float* token = (const float*)d_in[1];
    const float* W_c1  = (const float*)d_in[2];
    const float* b_c1  = (const float*)d_in[3];
    const float* W_c2  = (const float*)d_in[4];
    const float* b_c2  = (const float*)d_in[5];
    const float* W_s1  = (const float*)d_in[6];
    const float* b_s1  = (const float*)d_in[7];
    const float* W_s2  = (const float*)d_in[8];
    const float* b_s2  = (const float*)d_in[9];
    const float* W_t1  = (const float*)d_in[10];
    const float* b_t1  = (const float*)d_in[11];
    const float* W_t2  = (const float*)d_in[12];
    const float* b_t2  = (const float*)d_in[13];
    const float* a_par = (const float*)d_in[14];
    const float* b_par = (const float*)d_in[15];
    const float* p_par = (const float*)d_in[16];
    float* out = (float*)d_out;

    // ---- workspace carve-up ----
    char* base = (char*)d_ws;
    const size_t XT_BYTES = (size_t)BB * NN * CC * 2;  // 96 MiB
    const size_t HT_BYTES = (size_t)BB * NN * M1 * 2;  // 64 MiB
    unsigned short* xT = (unsigned short*)base;
    unsigned short* hT = (unsigned short*)(base + XT_BYTES);
    char* wr = base + XT_BYTES + HT_BYTES;             // persistent small region
    unsigned short* W1b   = (unsigned short*)wr;                 wr += (size_t)M1 * CC * 2;
    float*          bias1 = (float*)wr;                          wr += M1 * 4;
    unsigned short* Wc2b  = (unsigned short*)wr;                 wr += DD * 512 * 2;
    unsigned short* Ws2b  = (unsigned short*)wr;                 wr += KK * 512 * 2;
    float*          h1buf = (float*)wr;                          wr += (size_t)BB * 512 * 4;
    float*          tbuf  = (float*)wr;                          wr += (size_t)BB * 256 * 4;
    // alias region: lives inside xT, only used after GEMM1 (xT is dead then)
    char* al = base;
    float*          f_nd   = (float*)al;           al += (size_t)BB * NN * DD * 4;   // 16 MiB
    unsigned short* f_dn   = (unsigned short*)al;  al += (size_t)BB * DD * NN * 2;   //  8 MiB
    unsigned short* fnorm  = (unsigned short*)al;  al += (size_t)BB * NN * DD * 2;   //  8 MiB
    float*          scores = (float*)al;           al += (size_t)BB * NN * KK * 4;   //  8 MiB
    unsigned short* pmat   = (unsigned short*)al;  al += (size_t)BB * KK * NN * 2;   //  4 MiB
    float*          vbuf   = (float*)al;           al += (size_t)BB * DD * KK * 4;   //  1 MiB
    float*          wbuf   = (float*)al;           al += (size_t)BB * NN * 4;

    // 1) convert + transpose x ; pack weights
    convert_x_kernel<<<dim3(CC / 32, NN / 32, BB), 256, 0, stream>>>(f_map, xT);
    pack_w1_kernel<<<(M1 * CC + 255) / 256, 256, 0, stream>>>(W_c1, W_s1, W1b);
    pack_b1_kernel<<<(M1 + 255) / 256, 256, 0, stream>>>(b_c1, b_s1, bias1);
    f32_to_bf16_kernel<<<(DD * 512 + 255) / 256, 256, 0, stream>>>(W_c2, Wc2b, DD * 512);
    f32_to_bf16_kernel<<<(KK * 512 + 255) / 256, 256, 0, stream>>>(W_s2, Ws2b, KK * 512);

    // 2) GEMM1: hT[b][n][1024] = relu(W1 @ x + b1), bf16 out
    wmma_gemm_kernel<<<dim3(M1 / 64, NN / 256, BB), 256, 0, stream>>>(
        W1b, 0ull, CC, bias1,
        xT, (unsigned long long)NN * CC, CC, 0, CC, 1, NN,
        nullptr, 0ull, 0,
        hT, (unsigned long long)NN * M1, M1,
        nullptr, 0ull, 0);

    // 3) GEMM2a: f = W_c2 @ h + b_c2 -> f_nd (f32 [n][128]) and f_dn (bf16 [d][n])
    wmma_gemm_kernel<<<dim3(DD / 64, NN / 256, BB), 256, 0, stream>>>(
        Wc2b, 0ull, 512, b_c2,
        hT, (unsigned long long)NN * M1, M1, 0, 512, 0, NN,
        f_nd, (unsigned long long)NN * DD, DD,
        nullptr, 0ull, 0,
        f_dn, (unsigned long long)DD * NN, NN);

    // 4) GEMM2b: scores = W_s2 @ hs + b_s2 -> f32 [n][64]
    wmma_gemm_kernel<<<dim3(1, NN / 256, BB), 256, 0, stream>>>(
        Ws2b, 0ull, 512, b_s2,
        hT, (unsigned long long)NN * M1, M1, 512, 512, 0, NN,
        scores, (unsigned long long)NN * KK, KK,
        nullptr, 0ull, 0,
        nullptr, 0ull, 0);

    // 5) normalize f per pixel -> fnorm bf16
    normf_kernel<<<(BB * NN) / 256, 256, 0, stream>>>(f_nd, fnorm);

    // 6) fused sim GEMM + sigmoid + row reduction -> wbuf
    sim_kernel<<<(BB * (NN / 16)) / 8, 256, 0, stream>>>(fnorm, wbuf, a_par, b_par);

    // 7) softmax / w^p -> pmat bf16 [b][k][n]
    pmat_kernel<<<(BB * NN) / 256, 256, 0, stream>>>(scores, wbuf, p_par, pmat);

    // 8) VLAD: v[b][d][k] = sum_n f[d][n] pmat[k][n]  (A=pmat 64x1024, B=f_dn, transp. store)
    //    N'=128 < block tile 256 -> Nlimit guard retires the upper waves.
    wmma_gemm_kernel<<<dim3(1, 1, BB), 256, 0, stream>>>(
        pmat, (unsigned long long)KK * NN, NN, nullptr,
        f_dn, (unsigned long long)DD * NN, NN, 0, NN, 0, DD,
        vbuf, (unsigned long long)DD * KK, KK,
        nullptr, 0ull, 0,
        nullptr, 0ull, 0);

    // 9) token branch
    tok1_kernel<<<dim3(BB, 2), 256, 0, stream>>>(token, W_t1, b_t1, h1buf);
    tok2_kernel<<<BB, 256, 0, stream>>>(h1buf, W_t2, b_t2, tbuf);

    // 10) intra-norm + concat + global norm
    final_kernel<<<BB, 256, 0, stream>>>(vbuf, tbuf, out);
}
